// GCN_GAT_Model_64639257805287
// MI455X (gfx1250) — matrix operand order repaired and checked by reading, not compiled, not run on hardware
//
#include <hip/hip_runtime.h>
#include <hip/hip_bf16.h>

// ---------------------------------------------------------------------------
// GAT-style GNN on MI455X (gfx1250): bf16 WMMA GEMMs + atomic graph ops.
// ---------------------------------------------------------------------------

typedef __attribute__((ext_vector_type(16))) __bf16         v16bf;
typedef __attribute__((ext_vector_type(8)))  float          v8f;
typedef __attribute__((ext_vector_type(8)))  unsigned short us8;

union FragBF {
    v16bf v;
    us8   h[2];
    unsigned short u[16];
};

__device__ __forceinline__ unsigned short f2bf(float f) {
    unsigned int u = __float_as_uint(f);
    unsigned int r = u + 0x7FFFu + ((u >> 16) & 1u);   // round-to-nearest-even
    return (unsigned short)(r >> 16);
}

// float atomic max via ordered-int encoding (works for any sign mix)
__device__ __forceinline__ void atomicMaxF(float* addr, float v) {
    if (v >= 0.0f) atomicMax((int*)addr, __float_as_int(v));
    else           atomicMin((unsigned int*)addr, __float_as_uint(v));
}

// ---------------------------------------------------------------------------
// C[M, NT*16] = A[M,K](f32, K mult of 32, dense) @ W[NT*16, K](bf16)^T
// optional: + bias[col], + resid[row,col], relu. W staged in LDS.
// 256 threads = 8 waves; wave w owns rows [blk*128 + w*16, +16), all N tiles.
// ---------------------------------------------------------------------------
template <int NT>
__global__ __launch_bounds__(256) void wmma_gemm(
    const float* __restrict__ A, const unsigned short* __restrict__ Wb,
    const float* __restrict__ bias, const float* __restrict__ resid,
    float* __restrict__ C, int M, int K, int relu)
{
    constexpr int NOUT = NT * 16;
    extern __shared__ uint4 smem_raw[];
    unsigned short* smem = (unsigned short*)smem_raw;

    const int tid  = threadIdx.x;
    const int wave = tid >> 5;
    const int lane = tid & 31;

    { // cooperative stage of W (bf16, row-major [NOUT,K]) into LDS
        const int nvec = (NOUT * K) >> 3;
        const uint4* wsrc = (const uint4*)Wb;
        for (int i = tid; i < nvec; i += 256) smem_raw[i] = wsrc[i];
    }
    __syncthreads();

    const int rowBase = blockIdx.x * 128 + wave * 16;
    const int r       = rowBase + (lane & 15);
    const bool rowOK  = (r < M);
    const float* __restrict__ arow = A + (size_t)(rowOK ? r : 0) * K;

    v8f acc[NT];
#pragma unroll
    for (int nt = 0; nt < NT; ++nt)
#pragma unroll
        for (int i = 0; i < 8; ++i) acc[nt][i] = 0.0f;

    const int ah = (lane >> 4) * 8;    // A half-lane K sub-base (0 / 8)
    const int bh = (lane >> 4) * 8;    // B half-lane K sub-base (0 / 8)
    const unsigned short* wbase = smem + (size_t)(lane & 15) * K;

    for (int kc = 0; kc < K; kc += 32) {
        // speculative prefetch of next A chunk (safe past end: silently dropped)
        __builtin_prefetch(arow + kc + 32, 0, 2);

        // ---- A fragment: row r, K pattern {k0..k0+7, k0+16..k0+23} ----
        FragBF a;
        const int k0 = kc + ah;
        const int k1 = k0 + 16;
        float4 f0 = *(const float4*)(arow + k0);
        float4 f1 = *(const float4*)(arow + k0 + 4);
        float4 f2 = *(const float4*)(arow + k1);
        float4 f3 = *(const float4*)(arow + k1 + 4);
        a.u[0]  = f2bf(f0.x); a.u[1]  = f2bf(f0.y); a.u[2]  = f2bf(f0.z); a.u[3]  = f2bf(f0.w);
        a.u[4]  = f2bf(f1.x); a.u[5]  = f2bf(f1.y); a.u[6]  = f2bf(f1.z); a.u[7]  = f2bf(f1.w);
        a.u[8]  = f2bf(f2.x); a.u[9]  = f2bf(f2.y); a.u[10] = f2bf(f2.z); a.u[11] = f2bf(f2.w);
        a.u[12] = f2bf(f3.x); a.u[13] = f2bf(f3.y); a.u[14] = f2bf(f3.z); a.u[15] = f2bf(f3.w);

        // ---- B fragments (W row n = nt*16 + lane&15, K pattern {k0..k0+7, k0+16..k0+23}) ----
#pragma unroll
        for (int nt = 0; nt < NT; ++nt) {
            FragBF b;
            const unsigned short* wr = wbase + (size_t)nt * 16 * K + kc + bh;
            b.h[0] = *(const us8*)(wr);
            b.h[1] = *(const us8*)(wr + 16);
            acc[nt] = __builtin_amdgcn_wmma_f32_16x16x32_bf16(
                false, a.v, false, b.v, (short)0, acc[nt], false, false);
        }
    }

    // ---- epilogue + store: row = rowBase + i + 8*(lane>>4), col = lane&15 ----
    const int rsub = 8 * (lane >> 4);
#pragma unroll
    for (int nt = 0; nt < NT; ++nt) {
        const int col = nt * 16 + (lane & 15);
        const float bv = bias ? bias[col] : 0.0f;
#pragma unroll
        for (int i = 0; i < 8; ++i) {
            const int row = rowBase + rsub + i;
            if (row < M) {
                float cv = acc[nt][i] + bv;
                if (resid) cv += resid[(size_t)row * NOUT + col];
                if (relu)  cv = fmaxf(cv, 0.0f);
                C[(size_t)row * NOUT + col] = cv;
            }
        }
    }
}

// ---------------------------------------------------------------------------
// helpers
// ---------------------------------------------------------------------------
__global__ void cast_pad_bf16(const float* __restrict__ src, unsigned short* __restrict__ dst,
                              int rows, int kreal, int kpad) {
    int i = blockIdx.x * 256 + threadIdx.x;
    if (i >= rows * kpad) return;
    int r = i / kpad, k = i - r * kpad;
    float x = (k < kreal) ? src[r * kreal + k] : 0.0f;
    dst[i] = f2bf(x);
}

// zero-padded f32 copy: dst[r,kpad] = src[r,kreal] (k>=kreal -> 0)
__global__ void pad_rows_f32(const float* __restrict__ src, float* __restrict__ dst,
                             int rows, int kreal, int kpad) {
    int i = blockIdx.x * 256 + threadIdx.x;
    if (i >= rows * kpad) return;
    int r = i / kpad, k = i - r * kpad;
    dst[i] = (k < kreal) ? src[r * kreal + k] : 0.0f;
}

__global__ void fill_f32(float* __restrict__ p, float v, int n) {
    int i = blockIdx.x * 256 + threadIdx.x;
    if (i < n) p[i] = v;
}

// attn = u[src]*(v[dst]+x3)*D^-0.5 ; amax[dst] = max
__global__ void edge_attn_max(const float* __restrict__ u, const float* __restrict__ v,
                              const float* __restrict__ x3, const int* __restrict__ src,
                              const int* __restrict__ dst, float* __restrict__ attn,
                              float* __restrict__ amax, int E) {
    int i = blockIdx.x * 256 + threadIdx.x;
    if (i >= E * 128) return;
    int e = i >> 7, c = i & 127;
    int s = src[e], d = dst[e];
    float a = u[s * 128 + c] * (v[d * 128 + c] + x3[i]) * 0.17677669529663687f;
    attn[i] = a;
    atomicMaxF(&amax[d * 128 + c], a);
}

// ex = exp(attn - amax[dst]); den[dst] += ex; attn <- ex
__global__ void edge_exp_sum(float* __restrict__ attn, const float* __restrict__ amax,
                             const int* __restrict__ dst, float* __restrict__ den, int E) {
    int i = blockIdx.x * 256 + threadIdx.x;
    if (i >= E * 128) return;
    int e = i >> 7, c = i & 127;
    int d = dst[e];
    float ex = expf(attn[i] - amax[d * 128 + c]);
    attn[i] = ex;
    atomicAdd(&den[d * 128 + c], ex);
}

// agg[dst] += (ex/den[dst]) * (k4[dst] + x5)
__global__ void edge_apply(const float* __restrict__ attn, const float* __restrict__ den,
                           const float* __restrict__ k4, const float* __restrict__ x5,
                           const int* __restrict__ dst, float* __restrict__ agg, int E) {
    int i = blockIdx.x * 256 + threadIdx.x;
    if (i >= E * 128) return;
    int e = i >> 7, c = i & 127;
    int d = dst[e];
    float w = attn[i] / den[d * 128 + c];
    atomicAdd(&agg[d * 128 + c], w * (k4[d * 128 + c] + x5[i]));
}

// LayerNorm over 128 channels, one wave32 per row. out = LN(a (+ b))
__global__ __launch_bounds__(256) void ln_kernel(const float* __restrict__ a,
                                                 const float* __restrict__ b,
                                                 const float* __restrict__ gamma,
                                                 const float* __restrict__ beta,
                                                 float* __restrict__ out, int n) {
    int node = (blockIdx.x * 256 + threadIdx.x) >> 5;
    int lane = threadIdx.x & 31;
    if (node >= n) return;
    size_t base = (size_t)node * 128;
    float4 x = ((const float4*)(a + base))[lane];
    if (b) {
        float4 y = ((const float4*)(b + base))[lane];
        x.x += y.x; x.y += y.y; x.z += y.z; x.w += y.w;
    }
    float s = x.x + x.y + x.z + x.w;
#pragma unroll
    for (int m = 16; m > 0; m >>= 1) s += __shfl_xor(s, m, 32);
    float mu = s * (1.0f / 128.0f);
    float dx = x.x - mu, dy = x.y - mu, dz = x.z - mu, dw = x.w - mu;
    float q = dx * dx + dy * dy + dz * dz + dw * dw;
#pragma unroll
    for (int m = 16; m > 0; m >>= 1) q += __shfl_xor(q, m, 32);
    float inv = rsqrtf(q * (1.0f / 128.0f) + 1e-5f);
    float4 g  = ((const float4*)gamma)[lane];
    float4 be = ((const float4*)beta)[lane];
    float4 o;
    o.x = dx * inv * g.x + be.x;
    o.y = dy * inv * g.y + be.y;
    o.z = dz * inv * g.z + be.z;
    o.w = dw * inv * g.w + be.w;
    ((float4*)(out + base))[lane] = o;
}

__global__ void readout_scatter(const float* __restrict__ h, const int* __restrict__ gid,
                                float* __restrict__ g, int N) {
    int i = blockIdx.x * 256 + threadIdx.x;
    if (i >= N * 128) return;
    int n = i >> 7, c = i & 127;
    atomicAdd(&g[gid[n] * 128 + c], h[i]);
}

__global__ __launch_bounds__(256) void readout_dot(const float* __restrict__ g,
                                                   const float* __restrict__ w,
                                                   float* __restrict__ out, int G) {
    int gr = (blockIdx.x * 256 + threadIdx.x) >> 5;
    int lane = threadIdx.x & 31;
    if (gr >= G) return;
    float4 a = ((const float4*)(g + (size_t)gr * 128))[lane];
    float4 b = ((const float4*)w)[lane];
    float s = a.x * b.x + a.y * b.y + a.z * b.z + a.w * b.w;
#pragma unroll
    for (int m = 16; m > 0; m >>= 1) s += __shfl_xor(s, m, 32);
    if (lane == 0) out[gr] = s;
}

// ---------------------------------------------------------------------------
// host orchestration
// ---------------------------------------------------------------------------
extern "C" void kernel_launch(void* const* d_in, const int* in_sizes, int n_in,
                              void* d_out, int out_size, void* d_ws, size_t ws_size,
                              hipStream_t stream) {
    (void)in_sizes; (void)n_in; (void)out_size; (void)ws_size;
    const int N = 80000, E = 400000, G = 2048;

    const float* node_feats = (const float*)d_in[0];
    const float* edge_feats = (const float*)d_in[1];
    const int*   src        = (const int*)d_in[2];
    const int*   dst        = (const int*)d_in[3];
    const int*   gid        = (const int*)d_in[4];
    const float* emb_node   = (const float*)d_in[5];
    const float* emb_edge   = (const float*)d_in[6];
    const float* w_out      = (const float*)d_in[7];

    char* ws = (char*)d_ws;
    size_t off = 0;
    auto take = [&](size_t bytes) -> char* {
        char* p = ws + off;
        off = (off + bytes + 255) & ~(size_t)255;
        return p;
    };

    // bf16 weight staging buffers
    unsigned short* wb_embn = (unsigned short*)take(128 * 64 * 2);
    unsigned short* wb_embe = (unsigned short*)take(128 * 32 * 2);
    unsigned short* wb_w[4][6];
    unsigned short* wb_mw1[4];
    unsigned short* wb_mw2[4];
    for (int l = 0; l < 4; ++l) {
        for (int j = 0; j < 6; ++j) wb_w[l][j] = (unsigned short*)take(128 * 128 * 2);
        wb_mw1[l] = (unsigned short*)take(256 * 128 * 2);
        wb_mw2[l] = (unsigned short*)take(128 * 256 * 2);
    }

    // activation buffers
    float* nf_pad = (float*)take((size_t)N * 64 * 4);   // node_feats zero-padded K=59->64
    float* ef_pad = (float*)take((size_t)E * 32 * 4);   // edge_feats zero-padded K=6->32
    float* h    = (float*)take((size_t)N * 128 * 4);
    float* hln  = (float*)take((size_t)N * 128 * 4);
    float* htmp = (float*)take((size_t)N * 128 * 4);
    float* u    = (float*)take((size_t)N * 128 * 4);
    float* vv   = (float*)take((size_t)N * 128 * 4);
    float* kk   = (float*)take((size_t)N * 128 * 4);
    float* agg  = (float*)take((size_t)N * 128 * 4);
    float* amax = (float*)take((size_t)N * 128 * 4);
    float* den  = (float*)take((size_t)N * 128 * 4);
    float* ebuf = (float*)take((size_t)E * 128 * 4);
    float* x3   = (float*)take((size_t)E * 128 * 4);
    float* x5   = (float*)take((size_t)E * 128 * 4);
    float* attn = (float*)take((size_t)E * 128 * 4);
    float* t256 = attn; // alias: attn dead once MLP starts, rebuilt next layer
    float* gsum = (float*)take((size_t)G * 128 * 4);

    auto cast = [&](const float* s, unsigned short* d2, int rows, int kreal, int kpad) {
        int n = rows * kpad;
        cast_pad_bf16<<<(n + 255) / 256, 256, 0, stream>>>(s, d2, rows, kreal, kpad);
    };
    auto gemm128 = [&](const float* A, const unsigned short* W, const float* bias,
                       const float* resid, float* C, int M, int K, int relu) {
        wmma_gemm<8><<<(M + 127) / 128, 256, (size_t)128 * K * 2, stream>>>(
            A, W, bias, resid, C, M, K, relu);
    };
    auto gemm256 = [&](const float* A, const unsigned short* W, const float* bias,
                       const float* resid, float* C, int M, int K, int relu) {
        wmma_gemm<16><<<(M + 127) / 128, 256, (size_t)256 * K * 2, stream>>>(
            A, W, bias, resid, C, M, K, relu);
    };
    auto fill = [&](float* p, float v, int n) {
        fill_f32<<<(n + 255) / 256, 256, 0, stream>>>(p, v, n);
    };

    // -------- weight casts (bf16, K zero-padded to multiple of 32) --------
    cast(emb_node, wb_embn, 128, 59, 64);
    cast(emb_edge, wb_embe, 128, 6, 32);
    for (int l = 0; l < 4; ++l) {
        int base = 8 + 12 * l;
        for (int j = 0; j < 6; ++j)
            cast((const float*)d_in[base + j], wb_w[l][j], 128, 128, 128);
        cast((const float*)d_in[base + 8], wb_mw1[l], 256, 128, 128);   // mw1 [256,128]
        cast((const float*)d_in[base + 10], wb_mw2[l], 128, 256, 256);  // mw2 [128,256]
    }

    // -------- zero-pad raw features so every GEMM runs the dense path --------
    pad_rows_f32<<<((N * 64) + 255) / 256, 256, 0, stream>>>(node_feats, nf_pad, N, 59, 64);
    pad_rows_f32<<<((E * 32) + 255) / 256, 256, 0, stream>>>(edge_feats, ef_pad, E, 6, 32);

    // -------- embeddings --------
    gemm128(nf_pad, wb_embn, nullptr, nullptr, h, N, 64, 0);
    gemm128(ef_pad, wb_embe, nullptr, nullptr, ebuf, E, 32, 0);

    // -------- GAT layers --------
    const int ne = E * 128, nn = N * 128;
    for (int l = 0; l < 4; ++l) {
        int base = 8 + 12 * l;
        const float* ln_g = (const float*)d_in[base + 6];
        const float* ln_b = (const float*)d_in[base + 7];
        const float* mb1  = (const float*)d_in[base + 9];
        const float* mb2  = (const float*)d_in[base + 11];

        gemm128(h, wb_w[l][0], nullptr, nullptr, u,    N, 128, 0); // w1 -> u
        gemm128(h, wb_w[l][1], nullptr, nullptr, vv,   N, 128, 0); // w2 -> v
        gemm128(h, wb_w[l][3], nullptr, nullptr, kk,   N, 128, 0); // w4 -> k
        gemm128(h, wb_w[l][5], nullptr, nullptr, htmp, N, 128, 0); // w6 -> h6
        gemm128(ebuf, wb_w[l][2], nullptr, nullptr, x3, E, 128, 0); // w3
        gemm128(ebuf, wb_w[l][4], nullptr, nullptr, x5, E, 128, 0); // w5

        fill(amax, -3.0e38f, nn);
        fill(den, 0.0f, nn);
        fill(agg, 0.0f, nn);

        edge_attn_max<<<(ne + 255) / 256, 256, 0, stream>>>(u, vv, x3, src, dst, attn, amax, E);
        edge_exp_sum <<<(ne + 255) / 256, 256, 0, stream>>>(attn, amax, dst, den, E);
        edge_apply   <<<(ne + 255) / 256, 256, 0, stream>>>(attn, den, kk, x5, dst, agg, E);

        ln_kernel<<<(N * 32 + 255) / 256, 256, 0, stream>>>(htmp, agg, ln_g, ln_b, hln, N);

        // MLP: relu(hln@mw1^T + mb1) @ mw2^T + mb2 + hln, then LN -> h
        gemm256(hln, wb_mw1[l], mb1, nullptr, t256, N, 128, 1);
        gemm128(t256, wb_mw2[l], mb2, hln, htmp, N, 256, 0);
        ln_kernel<<<(N * 32 + 255) / 256, 256, 0, stream>>>(htmp, nullptr, ln_g, ln_b, h, N);
    }

    // -------- readout --------
    fill(gsum, 0.0f, G * 128);
    readout_scatter<<<(nn + 255) / 256, 256, 0, stream>>>(h, gid, gsum, N);
    readout_dot<<<(G * 32 + 255) / 256, 256, 0, stream>>>(gsum, w_out, (float*)d_out, G);
}
